// AttentionDecoderRNN_60086592471259
// MI455X (gfx1250) — compile-verified, matrix operand order checked by reading, not run on hardware
//
#include <hip/hip_runtime.h>

// Problem constants (from the reference)
#define VV   10000
#define EE   512
#define HH   512
#define ENCD 2048
#define AA   512
#define BB   128
#define PP   49
#define LL   25
#define TT   24
#define KCAT 3072   // E + ENC + H  (x = [emb | context | h])

typedef __attribute__((ext_vector_type(16))) __bf16 v16bf;
typedef __attribute__((ext_vector_type(8)))  __bf16 v8bf;
typedef __attribute__((ext_vector_type(8)))  float  v8f;

// ---- helpers -------------------------------------------------------------

__device__ __forceinline__ __bf16 f2bf(float f) {
    unsigned u = __builtin_bit_cast(unsigned, f);
    unsigned r = (u + 0x7FFFu + ((u >> 16) & 1u)) >> 16;   // round-nearest-even
    return __builtin_bit_cast(__bf16, (unsigned short)r);
}

__device__ __forceinline__ float wave_reduce_sum(float v) {
    #pragma unroll
    for (int off = 16; off > 0; off >>= 1) v += __shfl_xor(v, off, 32);
    return v;
}

__device__ __forceinline__ float sigmoidf(float x) {
    return 1.0f / (1.0f + __expf(-x));
}

// Load one 16x32 bf16 fragment (A: row-major rows = lanes; B: W row n = column n).
// Lane l<16 holds row/col (base+l), K = {k0..k0+7, k0+16..k0+23};
// lane l>=16 holds row/col (base+l-16), K = {k0+8..k0+15, k0+24..k0+31}.
__device__ __forceinline__ v16bf load_frag(const __bf16* __restrict__ base, int ld,
                                           int rc0, int k0, int lane) {
    int l16 = lane & 15;
    int kh  = (lane >> 4) << 3;     // 0 or 8
    const __bf16* p = base + (size_t)(rc0 + l16) * ld + k0 + kh;
    v8bf lo = *(const v8bf*)p;
    v8bf hi = *(const v8bf*)(p + 16);
    v16bf r;
    #pragma unroll
    for (int i = 0; i < 8; ++i) { r[i] = lo[i]; r[i + 8] = hi[i]; }
    return r;
}

// ---- setup kernels -------------------------------------------------------

__global__ void k_cvt_bf16(const float* __restrict__ src, __bf16* __restrict__ dst, int n) {
    int i = blockIdx.x * blockDim.x + threadIdx.x;
    if (i < n) dst[i] = f2bf(src[i]);
}

__global__ void k_zero_f32(float* __restrict__ p, int n) {
    int i = blockIdx.x * blockDim.x + threadIdx.x;
    if (i < n) p[i] = 0.0f;
}

// W_cat[2048][3072] = [ W_ih (2048x2560) | W_hh (2048x512) ]  (bf16)
__global__ void k_pack_wcat(const float* __restrict__ W_ih, const float* __restrict__ W_hh,
                            __bf16* __restrict__ W_cat) {
    int i = blockIdx.x * blockDim.x + threadIdx.x;
    if (i >= 4 * HH * KCAT) return;
    int row = i / KCAT, col = i % KCAT;
    float v = (col < (EE + ENCD)) ? W_ih[(size_t)row * (EE + ENCD) + col]
                                  : W_hh[(size_t)row * HH + (col - (EE + ENCD))];
    W_cat[i] = f2bf(v);
}

// emb_bf[b][t][e] = bf16(embed_table[captions[b][t]][e]), t in [0,T)
__global__ void k_embed(const float* __restrict__ table, const int* __restrict__ captions,
                        __bf16* __restrict__ emb) {
    int i = blockIdx.x * blockDim.x + threadIdx.x;
    if (i >= BB * TT * EE) return;
    int b = i / (TT * EE);
    int r = i % (TT * EE);
    int t = r / EE, e = r % EE;
    int tok = captions[b * LL + t];
    emb[i] = f2bf(table[(size_t)tok * EE + e]);
}

__global__ void k_bias_gates(const float* __restrict__ b_ih, const float* __restrict__ b_hh,
                             float* __restrict__ b_gates) {
    int i = blockIdx.x * blockDim.x + threadIdx.x;
    if (i < 4 * HH) b_gates[i] = b_ih[i] + b_hh[i];
}

// ---- WMMA GEMM: C[M,N] = A[M,K] * W[N,K]^T + bias -----------------------
// One wave computes a 16x64 tile (1 A-frag reused across 4 WMMAs per k-step).
// All tile indices are forced into SGPRs via readfirstlane so control flow
// around WMMA is scalar branches only (EXEC stays all-ones; no spill-inducing
// exec-mask divergence). Ragged N tiles are clamped in the k-loop (redundant
// compute) and masked only at the store.
// mode 0: C[m*ldc + n] = acc + bias[n]
// mode 1: preds epilogue -> out[m*T*V + t*V + n] = masked(acc + bias[n])
__global__ void k_wmma_gemm(const __bf16* __restrict__ A, int lda,
                            const __bf16* __restrict__ W, int ldw,
                            const float* __restrict__ bias,
                            float* __restrict__ C, int ldc,
                            int M, int N, int K,
                            int mode, const int* __restrict__ lengths, int t) {
    int lane = threadIdx.x & 31;
    int wave = __builtin_amdgcn_readfirstlane((blockIdx.x * blockDim.x + threadIdx.x) >> 5);
    int ntiles  = N >> 4;
    int nchunks = (ntiles + 3) >> 2;
    int mtiles  = M >> 4;
    if (wave >= mtiles * nchunks) return;

    int mt = wave / nchunks;
    int nc = wave - mt * nchunks;
    int m0 = mt << 4;
    int n0base = nc << 6;

    // clamped (always-valid) column-tile origins — scalar values
    int n0c[4];
    #pragma unroll
    for (int j = 0; j < 4; ++j) {
        int n0 = n0base + (j << 4);
        n0c[j] = n0 < N ? n0 : (N - 16);
    }

    v8f acc[4] = {v8f{}, v8f{}, v8f{}, v8f{}};

    for (int k0 = 0; k0 < K; k0 += 32) {
        v16bf af = load_frag(A, lda, m0, k0, lane);
        #pragma unroll
        for (int j = 0; j < 4; ++j) {
            v16bf bf = load_frag(W, ldw, n0c[j], k0, lane);
            acc[j] = __builtin_amdgcn_wmma_f32_16x16x32_bf16(
                false, af, false, bf, (short)0, acc[j], false, false);
        }
    }

    int l16 = lane & 15;
    int mo  = (lane >> 4) << 3;
    #pragma unroll
    for (int j = 0; j < 4; ++j) {
        int n0 = n0base + (j << 4);
        if (n0 >= N) continue;          // scalar branch (n0base is SGPR)
        int n = n0 + l16;
        float bv = bias ? bias[n] : 0.0f;
        if (mode == 0) {
            #pragma unroll
            for (int r = 0; r < 8; ++r) {
                int m = m0 + mo + r;
                C[(size_t)m * ldc + n] = acc[j][r] + bv;
            }
        } else {
            #pragma unroll
            for (int r = 0; r < 8; ++r) {
                int m = m0 + mo + r;               // m == batch index
                float v = acc[j][r] + bv;
                if (lengths[m] <= t) v = 0.0f;
                C[(size_t)m * TT * VV + (size_t)t * VV + n] = v;
            }
        }
    }
}

// ---- per-step kernels ----------------------------------------------------

// h -> h_bf; x_cat[:,0:E] = emb_t ; x_cat[:,E+ENC:] = h_bf
__global__ void k_assemble(const float* __restrict__ h, const __bf16* __restrict__ emb,
                           __bf16* __restrict__ h_bf, __bf16* __restrict__ x_cat, int t) {
    int i = blockIdx.x * blockDim.x + threadIdx.x;
    if (i >= BB * HH) return;
    int b = i / HH, e = i % HH;
    __bf16 hv = f2bf(h[i]);
    h_bf[i] = hv;
    x_cat[(size_t)b * KCAT + (EE + ENCD) + e] = hv;
    x_cat[(size_t)b * KCAT + e] = emb[((size_t)b * TT + t) * EE + e];
}

// one block per batch element: scores + softmax + context (+ masked alpha out)
__global__ void k_attention(const float* __restrict__ enc_att, const float* __restrict__ dec_att,
                            const float* __restrict__ w_full, const float* __restrict__ b_full,
                            const float* __restrict__ encoder_out, const int* __restrict__ lengths,
                            int t, float* __restrict__ alphas_out, __bf16* __restrict__ x_cat) {
    int b = blockIdx.x;
    int tid = threadIdx.x, lane = tid & 31, wid = tid >> 5;
    __shared__ float sh[64];

    for (int p = wid; p < PP; p += 8) {
        float s = 0.0f;
        for (int a = lane; a < AA; a += 32) {
            float v = enc_att[((size_t)b * PP + p) * AA + a] + dec_att[(size_t)b * AA + a];
            v = v > 0.0f ? v : 0.0f;                  // relu
            s += v * w_full[a];
        }
        s = wave_reduce_sum(s);
        if (lane == 0) sh[p] = s + b_full[0];
    }
    __syncthreads();

    if (tid == 0) {
        float mx = -1e30f;
        for (int p = 0; p < PP; ++p) mx = fmaxf(mx, sh[p]);
        float den = 0.0f;
        for (int p = 0; p < PP; ++p) { float e = __expf(sh[p] - mx); sh[p] = e; den += e; }
        float inv = 1.0f / den;
        for (int p = 0; p < PP; ++p) sh[p] *= inv;
    }
    __syncthreads();

    bool active = lengths[b] > t;
    if (tid < PP)
        alphas_out[((size_t)b * TT + t) * PP + tid] = active ? sh[tid] : 0.0f;

    // context (uses UNMASKED alpha, matching the reference)
    for (int e = tid; e < ENCD; e += blockDim.x) {
        float s = 0.0f;
        #pragma unroll 7
        for (int p = 0; p < PP; ++p)
            s += sh[p] * encoder_out[((size_t)b * PP + p) * ENCD + e];
        x_cat[(size_t)b * KCAT + EE + e] = f2bf(s);
    }
}

// LSTM pointwise: gates -> (h,c) with masking; emit h_new (bf16) for preds GEMM
__global__ void k_lstm(const float* __restrict__ gates, const int* __restrict__ lengths, int t,
                       float* __restrict__ h, float* __restrict__ c,
                       __bf16* __restrict__ hnew_bf) {
    int i = blockIdx.x * blockDim.x + threadIdx.x;
    if (i >= BB * HH) return;
    int b = i / HH, j = i % HH;
    const float* g = gates + (size_t)b * 4 * HH;
    float ig = sigmoidf(g[j]);
    float fg = sigmoidf(g[HH + j]);
    float gg = tanhf(g[2 * HH + j]);
    float og = sigmoidf(g[3 * HH + j]);
    float c_new = fg * c[i] + ig * gg;
    float h_new = og * tanhf(c_new);
    bool active = lengths[b] > t;
    if (active) { h[i] = h_new; c[i] = c_new; }
    hnew_bf[i] = f2bf(h_new);   // preds always use h_new (reference semantics)
}

// ---- driver --------------------------------------------------------------

extern "C" void kernel_launch(void* const* d_in, const int* in_sizes, int n_in,
                              void* d_out, int out_size, void* d_ws, size_t ws_size,
                              hipStream_t stream) {
    const float* encoder_out = (const float*)d_in[0];
    const int*   captions    = (const int*)  d_in[1];
    const int*   lengths     = (const int*)  d_in[2];
    const float* embed_table = (const float*)d_in[3];
    const float* W_enc       = (const float*)d_in[4];
    const float* b_enc       = (const float*)d_in[5];
    const float* W_dec       = (const float*)d_in[6];
    const float* b_dec       = (const float*)d_in[7];
    const float* w_full      = (const float*)d_in[8];
    const float* b_full      = (const float*)d_in[9];
    const float* W_ih        = (const float*)d_in[10];
    const float* b_ih        = (const float*)d_in[11];
    const float* W_hh        = (const float*)d_in[12];
    const float* b_hh        = (const float*)d_in[13];
    const float* W_lin       = (const float*)d_in[14];
    const float* b_lin       = (const float*)d_in[15];

    float* out_preds  = (float*)d_out;                       // [B,T,V]
    float* out_alphas = out_preds + (size_t)BB * TT * VV;    // [B,T,P]

    // workspace carve
    char* ws = (char*)d_ws;
    size_t off = 0;
    auto alloc = [&](size_t bytes) -> void* {
        off = (off + 255) & ~(size_t)255;
        void* p = ws + off;
        off += bytes;
        return p;
    };
    __bf16* enc_bf    = (__bf16*)alloc((size_t)BB * PP * ENCD * 2);
    __bf16* W_enc_bf  = (__bf16*)alloc((size_t)AA * ENCD * 2);
    __bf16* W_dec_bf  = (__bf16*)alloc((size_t)AA * HH * 2);
    __bf16* W_cat_bf  = (__bf16*)alloc((size_t)4 * HH * KCAT * 2);
    __bf16* W_lin_bf  = (__bf16*)alloc((size_t)VV * HH * 2);
    __bf16* emb_bf    = (__bf16*)alloc((size_t)BB * TT * EE * 2);
    float*  enc_att   = (float*) alloc((size_t)BB * PP * AA * 4);
    float*  b_gates   = (float*) alloc((size_t)4 * HH * 4);
    __bf16* x_cat     = (__bf16*)alloc((size_t)BB * KCAT * 2);
    float*  h_st      = (float*) alloc((size_t)BB * HH * 4);
    float*  c_st      = (float*) alloc((size_t)BB * HH * 4);
    __bf16* h_bf      = (__bf16*)alloc((size_t)BB * HH * 2);
    __bf16* hnew_bf   = (__bf16*)alloc((size_t)BB * HH * 2);
    float*  dec_att   = (float*) alloc((size_t)BB * AA * 4);
    float*  gates     = (float*) alloc((size_t)BB * 4 * HH * 4);

    const int TPB = 256;
    auto blocks = [](size_t n, int tpb) { return (int)((n + tpb - 1) / tpb); };
    auto gemm_blocks = [](int M, int N) {
        int waves = (M >> 4) * (((N >> 4) + 3) >> 2);
        return (waves + 7) / 8;                       // 8 waves / 256-thread block
    };

    // ---- setup
    k_cvt_bf16<<<blocks((size_t)BB * PP * ENCD, TPB), TPB, 0, stream>>>(encoder_out, enc_bf, BB * PP * ENCD);
    k_cvt_bf16<<<blocks((size_t)AA * ENCD, TPB), TPB, 0, stream>>>(W_enc, W_enc_bf, AA * ENCD);
    k_cvt_bf16<<<blocks((size_t)AA * HH, TPB), TPB, 0, stream>>>(W_dec, W_dec_bf, AA * HH);
    k_cvt_bf16<<<blocks((size_t)VV * HH, TPB), TPB, 0, stream>>>(W_lin, W_lin_bf, VV * HH);
    k_pack_wcat<<<blocks((size_t)4 * HH * KCAT, TPB), TPB, 0, stream>>>(W_ih, W_hh, W_cat_bf);
    k_embed<<<blocks((size_t)BB * TT * EE, TPB), TPB, 0, stream>>>(embed_table, captions, emb_bf);
    k_bias_gates<<<blocks(4 * HH, TPB), TPB, 0, stream>>>(b_ih, b_hh, b_gates);
    k_zero_f32<<<blocks((size_t)BB * HH, TPB), TPB, 0, stream>>>(h_st, BB * HH);
    k_zero_f32<<<blocks((size_t)BB * HH, TPB), TPB, 0, stream>>>(c_st, BB * HH);

    // enc_att[B*P, A] = enc * W_enc^T + b_enc   (one time)
    k_wmma_gemm<<<gemm_blocks(BB * PP, AA), TPB, 0, stream>>>(
        enc_bf, ENCD, W_enc_bf, ENCD, b_enc, enc_att, AA,
        BB * PP, AA, ENCD, 0, nullptr, 0);

    // ---- timestep loop (sequential dependence through h, c)
    for (int t = 0; t < TT; ++t) {
        k_assemble<<<blocks((size_t)BB * HH, TPB), TPB, 0, stream>>>(h_st, emb_bf, h_bf, x_cat, t);

        // dec_att[B, A] = h * W_dec^T + b_dec
        k_wmma_gemm<<<gemm_blocks(BB, AA), TPB, 0, stream>>>(
            h_bf, HH, W_dec_bf, HH, b_dec, dec_att, AA,
            BB, AA, HH, 0, nullptr, 0);

        k_attention<<<BB, TPB, 0, stream>>>(enc_att, dec_att, w_full, b_full,
                                            encoder_out, lengths, t, out_alphas, x_cat);

        // gates[B, 4H] = x_cat * W_cat^T + (b_ih + b_hh)
        k_wmma_gemm<<<gemm_blocks(BB, 4 * HH), TPB, 0, stream>>>(
            x_cat, KCAT, W_cat_bf, KCAT, b_gates, gates, 4 * HH,
            BB, 4 * HH, KCAT, 0, nullptr, 0);

        k_lstm<<<blocks((size_t)BB * HH, TPB), TPB, 0, stream>>>(gates, lengths, t, h_st, c_st, hnew_bf);

        // preds = h_new * W_lin^T + b_lin, masked, strided store into d_out
        k_wmma_gemm<<<gemm_blocks(BB, VV), TPB, 0, stream>>>(
            hnew_bf, HH, W_lin_bf, HH, b_lin, out_preds, VV,
            BB, VV, HH, 1, lengths, t);
    }
}
